// ProposalEncodingLayerV2_37443524886583
// MI455X (gfx1250) — compile-verified, matrix operand order checked by reading, not run on hardware
//
#include <hip/hip_runtime.h>
#include <hip/hip_bf16.h>
#include <float.h>
#include <stdint.h>

// ---------------- problem constants ----------------
#define BB     8
#define NN     256
#define DIMC   256
#define INTER  64
#define POS_H  64
#define ATTN_H 256

typedef _Float16     v16h  __attribute__((ext_vector_type(16)));
typedef float        v8f   __attribute__((ext_vector_type(8)));
typedef unsigned int u32x4 __attribute__((ext_vector_type(4)));
typedef float        f32x4 __attribute__((ext_vector_type(4)));

static __device__ __forceinline__ v8f zero8() {
    v8f z = {0.f,0.f,0.f,0.f,0.f,0.f,0.f,0.f};
    return z;
}

// Load a 16x32 (f16) WMMA A/B fragment from a row-major [row][k] array (ld in halves).
// ISA 7.12.2: lanes 0-15 -> rows 0-15, K = 0..7 / 16..23; lanes 16-31 -> K = 8..15 / 24..31.
// All addresses are 16B aligned -> two b128 loads per fragment.
static __device__ __forceinline__ v16h load_frag16(const _Float16* __restrict__ base,
                                                   int row0, int k0, int ld) {
    const int lane = threadIdx.x & 31;
    const int r    = lane & 15;
    const int kb   = (lane & 16) ? 8 : 0;
    const _Float16* p = base + (size_t)(row0 + r) * ld + k0 + kb;
    union { v16h v; u32x4 q[2]; } f;
    f.q[0] = *reinterpret_cast<const u32x4*>(p);
    f.q[1] = *reinterpret_cast<const u32x4*>(p + 16);
    return f.v;
}

// ---------------- kernel 1: weight transpose/convert to f16 ----------------
// Contiguous f16 block in ws: w1t[h][d] (256x64) | w2t[d][h] (64x256) | pw2t[o][h] (64x64)
__global__ void prep_weights_kernel(const float* __restrict__ attn_w1,
                                    const float* __restrict__ attn_w2,
                                    const float* __restrict__ pos_w2,
                                    _Float16* __restrict__ wall) {
    const int total = ATTN_H * INTER + INTER * ATTN_H + INTER * POS_H;
    for (int idx = blockIdx.x * blockDim.x + threadIdx.x; idx < total;
         idx += gridDim.x * blockDim.x) {
        if (idx < ATTN_H * INTER) {
            int h = idx >> 6, d = idx & 63;
            wall[idx] = (_Float16)attn_w1[d * ATTN_H + h];
        } else if (idx < 2 * ATTN_H * INTER) {
            int e = idx - ATTN_H * INTER;
            int d = e >> 8, h = e & 255;
            wall[idx] = (_Float16)attn_w2[h * INTER + d];
        } else {
            int e = idx - 2 * ATTN_H * INTER;
            int o = e >> 6, h = e & 63;
            wall[idx] = (_Float16)pos_w2[h * INTER + o];
        }
    }
}

// ---------------- kernel 2: q/k/v projections (thin GEMMs, VALU) ----------------
__global__ void qkv_kernel(const float* __restrict__ x, const float* __restrict__ y,
                           const float* __restrict__ theta_w, const float* __restrict__ phi_w,
                           const float* __restrict__ g_w,
                           float* __restrict__ q, float* __restrict__ k, float* __restrict__ v) {
    __shared__ float sx[DIMC];
    __shared__ float sy[DIMC];
    const int bi = blockIdx.x;      // b*N + n
    const int t  = threadIdx.x;     // 0..63
#pragma unroll
    for (int r = 0; r < 4; ++r) {
        sx[t + 64 * r] = x[(size_t)bi * DIMC + t + 64 * r];
        sy[t + 64 * r] = y[(size_t)bi * DIMC + t + 64 * r];
    }
    __syncthreads();
    float aq = 0.f, ak = 0.f, av = 0.f;
    for (int c = 0; c < DIMC; ++c) {
        aq = fmaf(sx[c], theta_w[c * INTER + t], aq);
        ak = fmaf(sy[c], phi_w[c * INTER + t], ak);
        av = fmaf(sy[c], g_w[c * INTER + t], av);
    }
    q[(size_t)bi * INTER + t] = aq;
    k[(size_t)bi * INTER + t] = ak;
    v[(size_t)bi * INTER + t] = av;
}

// ---------------- kernel 3: main fused attention, one block per (b,i) ----------------
struct __align__(16) Smem {
    union {
        struct { _Float16 A[NN * INTER]; _Float16 H[NN * INTER]; } g;  // 64 KB
        float SIM[NN * INTER];                                          // 64 KB (alias)
    } u;
    alignas(16) _Float16 RP[NN * INTER];   // 32 KB
    alignas(16) _Float16 W[36864];         // 72 KB: w1t | w2t | pw2t
    float qrow[INTER];
    int   jmask[NN];
    float part[4][INTER];
    float colMax[INTER];
    float colRcp[INTER];
    float agg[INTER];
};

__global__ void __launch_bounds__(256)
attn_main_kernel(const float* __restrict__ x,
                 const float* __restrict__ x_pos, const float* __restrict__ y_pos,
                 const int* __restrict__ mask,
                 const float* __restrict__ pos_w1, const float* __restrict__ pos_b1,
                 const float* __restrict__ pos_b2,
                 const float* __restrict__ attn_b1, const float* __restrict__ attn_b2,
                 const float* __restrict__ out_w,
                 const float* __restrict__ q, const float* __restrict__ k,
                 const float* __restrict__ v,
                 const _Float16* __restrict__ wall,
                 float* __restrict__ out) {
    __shared__ Smem s;
    const int bi = blockIdx.x;            // b*N + i
    const int b  = bi >> 8;
    const int t  = threadIdx.x;
    const int wave = t >> 5;
    const int lane = t & 31;
    const int cn   = lane & 15;            // N (=j) coordinate of C/D fragment
    const int mhi  = (lane & 16) ? 8 : 0;  // M (=channel) high offset of C/D fragment

    const _Float16* sW1t  = s.W;              // [h][d], ld 64
    const _Float16* sW2t  = s.W + 16384;      // [d][h], ld 256
    const _Float16* sPw2t = s.W + 32768;      // [o][h], ld 64

    // -------- phase 0a: kick off async weight staging (73728 B = 4608 x b128) --------
    {
        const char* src = (const char*)wall;
#pragma unroll
        for (int it = 0; it < 18; ++it) {
            const int idx = t + 256 * it;                       // 0..4607
            unsigned lds = (unsigned)(uintptr_t)(&s.W[idx * 8]);
            unsigned long long ga = (unsigned long long)(uintptr_t)(src + (size_t)idx * 16);
            asm volatile("global_load_async_to_lds_b128 %0, %1, off"
                         :: "v"(lds), "v"(ga) : "memory");
        }
    }

    // -------- phase 0b: stage per-(b,i) and per-b data --------
    if (t < INTER) s.qrow[t] = q[(size_t)bi * INTER + t];
    s.jmask[t] = mask[b * NN + t];
    const int mi = mask[bi];               // uniform -> scalar load

    // -------- phase 1: pos-MLP layer 1 (K=3, VALU), one j-row per thread --------
    {
        const int j = t;
        const float xp0 = x_pos[(size_t)bi * 3 + 0];   // uniform -> scalar
        const float xp1 = x_pos[(size_t)bi * 3 + 1];
        const float xp2 = x_pos[(size_t)bi * 3 + 2];
        const float r0 = xp0 - y_pos[((size_t)b * NN + j) * 3 + 0];
        const float r1 = xp1 - y_pos[((size_t)b * NN + j) * 3 + 1];
        const float r2 = xp2 - y_pos[((size_t)b * NN + j) * 3 + 2];
        _Float16* hp = &s.u.g.H[j * POS_H];
        for (int hc = 0; hc < 8; ++hc) {
            union { u32x4 qv; _Float16 h[8]; } pk;
#pragma unroll
            for (int e = 0; e < 8; ++e) {
                const int h = hc * 8 + e;       // pos_w1/pos_b1 reads are uniform
                float hv = fmaf(r0, pos_w1[h],
                          fmaf(r1, pos_w1[POS_H + h],
                          fmaf(r2, pos_w1[2 * POS_H + h], pos_b1[h])));
                pk.h[e] = (_Float16)fmaxf(hv, 0.f);
            }
            *reinterpret_cast<u32x4*>(hp + hc * 8) = pk.qv;
        }
    }
    // weights must be resident before phase 2 reads sPw2t
    asm volatile("s_wait_asynccnt 0x0" ::: "memory");
    __syncthreads();

    // -------- phase 2: RP^T = Pw2^T @ hidden1^T (WMMA); packed f16 b128 stores --------
#pragma unroll
    for (int tt = 0; tt < 8; ++tt) {
        const int id = wave * 8 + tt;
        const int jt = id >> 2, dt = id & 3;    // N-tile = j, M-tile = d
        v8f acc = zero8();
#pragma unroll
        for (int kk = 0; kk < 2; ++kk) {
            v16h aM = load_frag16(sPw2t, dt * 16, kk * 32, POS_H);
            v16h bN = load_frag16(s.u.g.H, jt * 16, kk * 32, POS_H);
            acc = __builtin_amdgcn_wmma_f32_16x16x32_f16(false, aM, false, bN,
                                                         (short)0, acc, false, false);
        }
        const int j = jt * 16 + cn;
        const int dbase = dt * 16 + mhi;
        f32x4 pb0 = *reinterpret_cast<const f32x4*>(pos_b2 + dbase);
        f32x4 pb1 = *reinterpret_cast<const f32x4*>(pos_b2 + dbase + 4);
        union { u32x4 qv; _Float16 h[8]; } pk;
#pragma unroll
        for (int vv = 0; vv < 4; ++vv) pk.h[vv] = (_Float16)(acc[vv] + pb0[vv]);
#pragma unroll
        for (int vv = 4; vv < 8; ++vv) pk.h[vv] = (_Float16)(acc[vv] + pb1[vv - 4]);
        *reinterpret_cast<u32x4*>(&s.RP[j * INTER + dbase]) = pk.qv;
    }
    __syncthreads();

    // -------- phase 2b: A = q - k + RP (cooperative, coalesced k reads) --------
    {
#pragma unroll
        for (int it = 0; it < 16; ++it) {
            const int idx = t + 256 * it;        // float4 group, 0..4095
            const int j  = idx >> 4;
            const int d4 = (idx & 15) << 2;
            f32x4 kv = *reinterpret_cast<const f32x4*>(k + ((size_t)b * NN + j) * INTER + d4);
            const _Float16* rp = &s.RP[j * INTER + d4];
            _Float16* ap = &s.u.g.A[j * INTER + d4];
#pragma unroll
            for (int c = 0; c < 4; ++c) {
                ap[c] = (_Float16)(s.qrow[d4 + c] - kv[c] + (float)rp[c]);
            }
        }
    }
    __syncthreads();

    // -------- phase 3: strips: GEMM1' H^T = W1^T@A^T (+bias+relu), GEMM2' accumulate --------
    v8f accS[8];
#pragma unroll
    for (int tt = 0; tt < 8; ++tt) accS[tt] = zero8();

    for (int sgr = 0; sgr < 4; ++sgr) {           // 4 strips of 64 hidden channels
        // GEMM1': per tile M=h, N=j; store packed into H[j][h]
#pragma unroll
        for (int tt = 0; tt < 8; ++tt) {
            const int id = wave * 8 + tt;
            const int jt = id >> 2, ht = id & 3;
            v8f hAcc = zero8();
#pragma unroll
            for (int kk = 0; kk < 2; ++kk) {
                v16h aM = load_frag16(sW1t, sgr * 64 + ht * 16, kk * 32, INTER);
                v16h bN = load_frag16(s.u.g.A, jt * 16, kk * 32, INTER);
                hAcc = __builtin_amdgcn_wmma_f32_16x16x32_f16(false, aM, false, bN,
                                                              (short)0, hAcc, false, false);
            }
            const int j = jt * 16 + cn;
            const int hbase = ht * 16 + mhi;                 // local h within strip
            f32x4 b0 = *reinterpret_cast<const f32x4*>(attn_b1 + sgr * 64 + hbase);
            f32x4 b1v = *reinterpret_cast<const f32x4*>(attn_b1 + sgr * 64 + hbase + 4);
            union { u32x4 qv; _Float16 h[8]; } pk;
#pragma unroll
            for (int vv = 0; vv < 4; ++vv)
                pk.h[vv] = (_Float16)fmaxf(hAcc[vv] + b0[vv], 0.f);
#pragma unroll
            for (int vv = 4; vv < 8; ++vv)
                pk.h[vv] = (_Float16)fmaxf(hAcc[vv] + b1v[vv - 4], 0.f);
            *reinterpret_cast<u32x4*>(&s.u.g.H[j * INTER + hbase]) = pk.qv;
        }
        __syncthreads();
        // GEMM2': SIM^T += W2^T[:,strip] @ H^T
#pragma unroll
        for (int tt = 0; tt < 8; ++tt) {
            const int id = wave * 8 + tt;
            const int jt = id >> 2, dt = id & 3;
#pragma unroll
            for (int kk = 0; kk < 2; ++kk) {
                v16h aM = load_frag16(sW2t, dt * 16, sgr * 64 + kk * 32, ATTN_H);
                v16h bN = load_frag16(s.u.g.H, jt * 16, kk * 32, INTER);
                accS[tt] = __builtin_amdgcn_wmma_f32_16x16x32_f16(false, aM, false, bN,
                                                                  (short)0, accS[tt],
                                                                  false, false);
            }
        }
        __syncthreads();
    }

    // -------- phase 4: SIM + b2, mask (one check per tile), packed b128 stores --------
#pragma unroll
    for (int tt = 0; tt < 8; ++tt) {
        const int id = wave * 8 + tt;
        const int jt = id >> 2, dt = id & 3;
        const int j = jt * 16 + cn;
        const int dbase = dt * 16 + mhi;
        const bool ok = (mi != 0) && (s.jmask[j] != 0);
        f32x4 b0 = *reinterpret_cast<const f32x4*>(attn_b2 + dbase);
        f32x4 b1v = *reinterpret_cast<const f32x4*>(attn_b2 + dbase + 4);
        f32x4 lo, hi;
#pragma unroll
        for (int c = 0; c < 4; ++c) {
            lo[c] = ok ? (accS[tt][c] + b0[c]) : -FLT_MAX;
            hi[c] = ok ? (accS[tt][4 + c] + b1v[c]) : -FLT_MAX;
        }
        *reinterpret_cast<f32x4*>(&s.u.SIM[j * INTER + dbase])     = lo;
        *reinterpret_cast<f32x4*>(&s.u.SIM[j * INTER + dbase + 4]) = hi;
    }
    __syncthreads();

    // -------- phase 5: channel-wise masked softmax over j --------
    const int d = t & 63;
    const int grp = t >> 6;
    {
        float m = -FLT_MAX;
        for (int jj = 0; jj < 64; ++jj) {
            const int j = grp * 64 + jj;
            m = fmaxf(m, s.u.SIM[j * INTER + d]);
        }
        s.part[grp][d] = m;
    }
    __syncthreads();
    if (t < 64) {
        float m = fmaxf(fmaxf(s.part[0][t], s.part[1][t]),
                        fmaxf(s.part[2][t], s.part[3][t]));
        s.colMax[t] = m;
    }
    __syncthreads();
    {
        const float cm = s.colMax[d];
        float ssum = 0.f;
        for (int jj = 0; jj < 64; ++jj) {
            const int j = grp * 64 + jj;
            float e = __expf(s.u.SIM[j * INTER + d] - cm);
            s.u.SIM[j * INTER + d] = e;
            ssum += e;
        }
        s.part[grp][d] = ssum;
    }
    __syncthreads();
    if (t < 64) {
        float tot = s.part[0][t] + s.part[1][t] + s.part[2][t] + s.part[3][t];
        s.colRcp[t] = 1.f / tot;
    }
    __syncthreads();

    // -------- phase 5b: agg[d] = sum_j attn[j][d] * (v[b,j,d] + RP[j][d]) --------
    {
        const float rcp = s.colRcp[d];
        float ag = 0.f;
        for (int jj = 0; jj < 64; ++jj) {
            const int j = grp * 64 + jj;
            float attn = s.u.SIM[j * INTER + d] * rcp;
            float vf = v[((size_t)b * NN + j) * INTER + d] + (float)s.RP[j * INTER + d];
            ag = fmaf(attn, vf, ag);
        }
        s.part[grp][d] = ag;
    }
    __syncthreads();
    if (t < 64) {
        s.agg[t] = s.part[0][t] + s.part[1][t] + s.part[2][t] + s.part[3][t];
    }
    __syncthreads();

    // -------- phase 6: out = x + agg @ out_w --------
    {
        const int c = t;
        float o = x[(size_t)bi * DIMC + c];
        for (int dd = 0; dd < INTER; ++dd) {
            o = fmaf(s.agg[dd], out_w[dd * DIMC + c], o);
        }
        out[(size_t)bi * DIMC + c] = o;
    }
}

// ---------------- launcher ----------------
extern "C" void kernel_launch(void* const* d_in, const int* in_sizes, int n_in,
                              void* d_out, int out_size, void* d_ws, size_t ws_size,
                              hipStream_t stream) {
    const float* x       = (const float*)d_in[0];
    const float* y       = (const float*)d_in[1];
    const float* x_pos   = (const float*)d_in[2];
    const float* y_pos   = (const float*)d_in[3];
    const int*   mask    = (const int*)  d_in[4];
    const float* g_w     = (const float*)d_in[5];
    const float* theta_w = (const float*)d_in[6];
    const float* phi_w   = (const float*)d_in[7];
    const float* pos_w1  = (const float*)d_in[8];
    const float* pos_b1  = (const float*)d_in[9];
    const float* pos_w2  = (const float*)d_in[10];
    const float* pos_b2  = (const float*)d_in[11];
    const float* attn_w1 = (const float*)d_in[12];
    const float* attn_b1 = (const float*)d_in[13];
    const float* attn_w2 = (const float*)d_in[14];
    const float* attn_b2 = (const float*)d_in[15];
    const float* out_w   = (const float*)d_in[16];
    float* out = (float*)d_out;

    char* ws = (char*)d_ws;
    const size_t QKV_BYTES = (size_t)BB * NN * INTER * sizeof(float); // 512 KB each
    float* q = (float*)(ws);
    float* k = (float*)(ws + QKV_BYTES);
    float* v = (float*)(ws + 2 * QKV_BYTES);
    _Float16* wall = (_Float16*)(ws + 3 * QKV_BYTES);   // 73728 B contiguous

    prep_weights_kernel<<<64, 256, 0, stream>>>(attn_w1, attn_w2, pos_w2, wall);
    qkv_kernel<<<BB * NN, 64, 0, stream>>>(x, y, theta_w, phi_w, g_w, q, k, v);
    attn_main_kernel<<<BB * NN, 256, 0, stream>>>(
        x, x_pos, y_pos, mask, pos_w1, pos_b1, pos_b2, attn_b1, attn_b2, out_w,
        q, k, v, wall, out);
}
